// CustomMHA_81793357185885
// MI455X (gfx1250) — matrix operand order repaired and checked by reading, not compiled, not run on hardware
//
#include <hip/hip_runtime.h>
#include <hip/hip_bf16.h>
#include <stdint.h>

typedef __attribute__((ext_vector_type(16))) _Float16 v16h;
typedef __attribute__((ext_vector_type(8)))  _Float16 v8h;
typedef __attribute__((ext_vector_type(2)))  _Float16 v2h;
typedef __attribute__((ext_vector_type(8)))  float    v8f;
typedef __attribute__((ext_vector_type(4)))  float    v4f;
typedef __attribute__((ext_vector_type(4)))  unsigned int v4u;
typedef __attribute__((ext_vector_type(8)))  int      v8i;
typedef __attribute__((ext_vector_type(4)))  int      v4i;

#define DIM   1024
#define NHEAD 16
#define HDIM  64
#define BS    4
#define QLEN  2048
#define KLEN  2048

// A-matrix (16x32 f16): lane holds row M = lane%16; half i holds
// K = (i&7) + (i>>3)*16 + (lane>>4)*8   (ISA 7.12.2 table).
__device__ __forceinline__ int kmap(int i, int lane) {
    return (i & 7) + ((i >> 3) << 4) + ((lane >> 4) << 3);
}
// B-matrix (32x16 f16): lane holds column N = lane%16; half i holds K = kmap(i, lane)

__device__ __forceinline__ v8f wmma_f16(v16h a, v16h b, v8f c) {
    return __builtin_amdgcn_wmma_f32_16x16x32_f16(
        false, a, false, b, (short)0, c, false, false);
}

__device__ __forceinline__ v16h cat8(v8h lo, v8h hi) {
    return __builtin_shufflevector(lo, hi, 0, 1, 2, 3, 4, 5, 6, 7,
                                           8, 9, 10, 11, 12, 13, 14, 15);
}

// gfx1250 async copy: 16 bytes global -> LDS, tracked by ASYNCcnt.
__device__ __forceinline__ void async_b128(uint32_t lds_byte, const void* gptr) {
    asm volatile("global_load_async_to_lds_b128 %0, %1, off"
                 :: "v"(lds_byte), "v"((unsigned long long)(uintptr_t)gptr)
                 : "memory");
}
__device__ __forceinline__ void wait_async0() {
    asm volatile("s_wait_asynccnt 0x0" ::: "memory");
}

// Tensor Data Mover: DMA a contiguous run of `nelem` f16 from global to LDS.
// D# per ISA 8.3/8.4: 1-D tile (tile_dim0 = nelem <= 65535, dim1 = 1),
// data_size = 2 bytes, no padding/iterate/multicast; groups 2/3 zero (<=2D).
// This toolchain exposes the 6-arg builtin (extra int32x8 group + cpol).
__device__ __forceinline__ void tdm_load_f16(uint32_t lds_byte, const void* gptr,
                                             int nelem) {
    unsigned long long ga = (unsigned long long)(uintptr_t)gptr;
    v4u g0;
    g0.x = 1u;                                        // count=1, user descriptor
    g0.y = lds_byte;                                  // lds_addr [63:32]
    g0.z = (unsigned)(ga & 0xffffffffu);              // global_addr[31:0]
    g0.w = (unsigned)((ga >> 32) & 0x01ffffffu)       // global_addr[56:32]
         | 0x80000000u;                               // type=2 ("image")
    v8i g1;
    g1[0] = (1 << 16);                                // data_size code 1 = 2B
    g1[1] = (nelem & 0xffff) << 16;                   // tensor_dim0[15:0]
    g1[2] = (int)((unsigned)nelem >> 16) | (1 << 16); // tensor_dim0[31:16], dim1=1
    g1[3] = (nelem & 0xffff) << 16;                   // tile_dim0
    g1[4] = 1;                                        // tile_dim1=1, tile_dim2=0
    g1[5] = nelem;                                    // tensor_dim0_stride[31:0]
    g1[6] = 0;
    g1[7] = 0;
    v4i z4 = {0, 0, 0, 0};
    v8i z8 = {0, 0, 0, 0, 0, 0, 0, 0};
    __builtin_amdgcn_tensor_load_to_lds(g0, g1, z4, z4, z8, 0);
}

// ---------------------------------------------------------------------------
// GEMM: Out(f16) = A(f32)[M,K] @ W(f32)[K,N] + bias, optional head-major remap
// Block: 8 waves -> 128 rows x 16 cols. W tile in LDS in B-fragment order [lane][16] f16:
// B-fragment per lane is one contiguous 32B read.
// ---------------------------------------------------------------------------
__global__ void __launch_bounds__(256)
qkv_proj_kernel(const float* __restrict__ A, const float* __restrict__ W,
                const float* __restrict__ bias, _Float16* __restrict__ Out,
                int M, int N, int K, int head_major)
{
    __shared__ _Float16 sW[32 * 16];   // [lane][16] B-fragment order
    const int tid  = threadIdx.x;
    const int wave = tid >> 5;
    const int lane = tid & 31;
    const int ntile = N >> 4;
    const int bn = blockIdx.x % ntile;
    const int bm = blockIdx.x / ntile;
    const int m0 = bm * 128 + wave * 16;
    const int n0 = bn * 16;
    const int r16   = lane & 15;
    const int rbase = (lane >> 4) << 3;

    const int sk  = tid >> 3;          // staging: k row 0..31
    const int sn2 = (tid & 7) * 2;     // n pair 0,2,..,14

    v8f acc = {};
    for (int kk = 0; kk < K; kk += 32) {
        __syncthreads();
        const float* wrow = W + (size_t)(kk + sk) * N + n0 + sn2;
        sW[(((sk >> 3) & 1) << 8) + sn2 * 16 + (sk & 7) + ((sk >> 4) << 3)]        = (_Float16)wrow[0];
        sW[(((sk >> 3) & 1) << 8) + (sn2 + 1) * 16 + (sk & 7) + ((sk >> 4) << 3)] = (_Float16)wrow[1];
        __syncthreads();

        v16h bF = *(const v16h*)(sW + lane * 16);   // 2x ds_load_b128

        const float* arow = A + (size_t)(m0 + r16) * K + kk;
        v4f a0 = *(const v4f*)(arow + rbase);
        v4f a1 = *(const v4f*)(arow + rbase + 4);
        v4f a2 = *(const v4f*)(arow + 16 + rbase);
        v4f a3 = *(const v4f*)(arow + 20 + rbase);
        v16h aF;
        #pragma unroll
        for (int i = 0; i < 4; ++i) {
            aF[i]      = (_Float16)a0[i];
            aF[4 + i]  = (_Float16)a1[i];
            aF[8 + i]  = (_Float16)a2[i];
            aF[12 + i] = (_Float16)a3[i];
        }
        if (kk + 64 <= K) __builtin_prefetch(arow + 32, 0, 3);
        acc = wmma_f16(aF, bF, acc);
    }

    float bv = bias[n0 + r16];
    #pragma unroll
    for (int r = 0; r < 8; ++r) {
        int m = m0 + r + rbase;
        int n = n0 + r16;
        float v = acc[r] + bv;
        size_t dst;
        if (head_major) {
            int b_ = m >> 11;          // QLEN = 2048 rows per batch
            int l  = m & 2047;
            int h  = n >> 6;           // HDIM = 64
            int dd = n & 63;
            dst = ((size_t)((b_ * NHEAD + h) * QLEN + l)) * HDIM + dd;
        } else {
            dst = (size_t)m * N + n;
        }
        Out[dst] = (_Float16)v;
    }
}

// ---------------------------------------------------------------------------
// GEMM: Out(f32) = A(f16)[M,K] @ W(f32)[K,N] + bias (output projection)
// ---------------------------------------------------------------------------
__global__ void __launch_bounds__(256)
out_proj_kernel(const _Float16* __restrict__ A, const float* __restrict__ W,
                const float* __restrict__ bias, float* __restrict__ Out,
                int M, int N, int K)
{
    __shared__ _Float16 sW[32 * 16];
    const int tid  = threadIdx.x;
    const int wave = tid >> 5;
    const int lane = tid & 31;
    const int ntile = N >> 4;
    const int bn = blockIdx.x % ntile;
    const int bm = blockIdx.x / ntile;
    const int m0 = bm * 128 + wave * 16;
    const int n0 = bn * 16;
    const int r16   = lane & 15;
    const int rbase = (lane >> 4) << 3;
    const int sk  = tid >> 3;
    const int sn2 = (tid & 7) * 2;

    v8f acc = {};
    for (int kk = 0; kk < K; kk += 32) {
        __syncthreads();
        const float* wrow = W + (size_t)(kk + sk) * N + n0 + sn2;
        sW[(((sk >> 3) & 1) << 8) + sn2 * 16 + (sk & 7) + ((sk >> 4) << 3)]        = (_Float16)wrow[0];
        sW[(((sk >> 3) & 1) << 8) + (sn2 + 1) * 16 + (sk & 7) + ((sk >> 4) << 3)] = (_Float16)wrow[1];
        __syncthreads();

        v16h bF = *(const v16h*)(sW + lane * 16);

        const _Float16* arow = A + (size_t)(m0 + r16) * K + kk;
        v16h aF = cat8(*(const v8h*)(arow + rbase),
                       *(const v8h*)(arow + 16 + rbase));
        if (kk + 64 <= K) __builtin_prefetch(arow + 32, 0, 3);
        acc = wmma_f16(aF, bF, acc);
    }

    float bv = bias[n0 + r16];
    #pragma unroll
    for (int r = 0; r < 8; ++r) {
        int m = m0 + r + rbase;
        int n = n0 + r16;
        Out[(size_t)m * N + n] = acc[r] + bv;
    }
}

// ---------------------------------------------------------------------------
// Block-cooperative flash attention, double-buffered DMA staging:
//   K tiles : Tensor Data Mover (tensor_load_to_lds, TENSORcnt) from wave 0
//   V tiles : per-lane async-to-LDS copies (ASYNCcnt)
// 8 waves share one (b,h); each owns a 16-row q-tile. K^T fragments come from
// LDS via plain ds_load_b128 of key rows; V fragments are natural row-major B reads.
// ---------------------------------------------------------------------------
__global__ void __launch_bounds__(256)
flash_attn_kernel(const _Float16* __restrict__ Qh, const _Float16* __restrict__ Kh,
                  const _Float16* __restrict__ Vh, const unsigned char* __restrict__ mask,
                  _Float16* __restrict__ Oh)
{
    __shared__ _Float16 sK[2][32 * HDIM];  // ping-pong 32 keys x 64 d, 4KB each
    __shared__ _Float16 sV[2][32 * HDIM];
    __shared__ _Float16 sP[8][32 * 16];    // per-wave P tile in A-fragment order

    const int tid  = threadIdx.x;
    const int wave = tid >> 5;
    const int lane = tid & 31;
    const int bh   = blockIdx.x >> 4;      // 64 (b,h) pairs
    const int qblk = blockIdx.x & 15;      // 16 q-blocks of 128 rows
    const int qt   = qblk * 8 + wave;      // q-tile 0..127
    const int b    = bh >> 4;
    const int h    = bh & 15;
    const _Float16* Qp = Qh + ((size_t)bh * QLEN + (size_t)qt * 16) * HDIM;
    const _Float16* Kp = Kh + (size_t)bh * KLEN * HDIM;
    const _Float16* Vp = Vh + (size_t)bh * KLEN * HDIM;
    const unsigned char* mp = mask + (size_t)b * KLEN;
    _Float16* sPw = sP[wave];
    const int r16   = lane & 15;
    const int rbase = (lane >> 4) << 3;

    const uint32_t ldsKb[2] = { (uint32_t)(uintptr_t)(&sK[0][0]),
                                (uint32_t)(uintptr_t)(&sK[1][0]) };
    const uint32_t ldsV[2]  = { (uint32_t)(uintptr_t)(&sV[0][0]) + tid * 16,
                                (uint32_t)(uintptr_t)(&sV[1][0]) + tid * 16 };

    // Q fragments (persist in VGPRs): A layout, d split into two K=32 chunks
    v16h qf[2];
    #pragma unroll
    for (int j = 0; j < 2; ++j) {
        const _Float16* qrow = Qp + (size_t)r16 * HDIM + j * 32;
        qf[j] = cat8(*(const v8h*)(qrow + rbase), *(const v8h*)(qrow + 16 + rbase));
    }

    v8f o0 = {}, o1 = {}, o2 = {}, o3 = {};
    float mrow[8], lrow[8];
    #pragma unroll
    for (int r = 0; r < 8; ++r) { mrow[r] = -1e30f; lrow[r] = 0.0f; }

    // prologue: kick off tile 0 (32x64 f16 tiles are fully contiguous)
    if (wave == 0) tdm_load_f16(ldsKb[0], Kp, 32 * HDIM);      // one TDM per tile
    async_b128(ldsV[0], (const char*)Vp + tid * 16);           // 256 x 16B slices

    int buf = 0;
    for (int kt = 0; kt < KLEN; kt += 32, buf ^= 1) {
        if (wave == 0) __builtin_amdgcn_s_wait_tensorcnt(0);   // K tile landed
        wait_async0();                                         // V slice landed
        __syncthreads();     // publish both tiles; prev compute done
        if (kt + 32 < KLEN) {   // overlap next tile's DMA with compute
            const size_t off = (size_t)(kt + 32) * HDIM;
            if (wave == 0) tdm_load_f16(ldsKb[buf ^ 1], Kp + off, 32 * HDIM);
            async_b128(ldsV[buf ^ 1], (const char*)(Vp + off) + tid * 16);
        }
        const _Float16* sKb = sK[buf];
        const _Float16* sVb = sV[buf];

        // --- S = Q K^T : B needs K^T (column-major in LDS) -> plain b128 row loads ---
        v8f s2[2];
        #pragma unroll
        for (int t = 0; t < 2; ++t) {
            v8f st = {};
            #pragma unroll
            for (int j = 0; j < 2; ++j) {
                const _Float16* base = sKb + (size_t)(t * 16 + r16) * HDIM + j * 32;
                v8h t0, t1;
                asm volatile("ds_load_b128 %0, %2\n\t"
                             "ds_load_b128 %1, %3"
                             : "=&v"(t0), "=&v"(t1)
                             : "v"((uint32_t)(uintptr_t)(base + rbase)),
                               "v"((uint32_t)(uintptr_t)(base + 16 + rbase)));
                asm volatile("s_wait_dscnt 0x0" ::: "memory");
                st = wmma_f16(qf[j], cat8(t0, t1), st);
            }
            const bool mk = mp[kt + t * 16 + r16] != 0;
            #pragma unroll
            for (int r = 0; r < 8; ++r)
                st[r] = mk ? -1e30f : st[r] * 0.125f;   // 1/sqrt(64)
            s2[t] = st;
        }

        // --- online softmax; P written in A-fragment order ---
        #pragma unroll
        for (int r = 0; r < 8; ++r) {
            float mx = fmaxf(s2[0][r], s2[1][r]);
            #pragma unroll
            for (int off = 1; off <= 8; off <<= 1)
                mx = fmaxf(mx, __shfl_xor(mx, off, 32));
            float mnew  = fmaxf(mrow[r], mx);
            float alpha = __expf(mrow[r] - mnew);
            float p0 = __expf(s2[0][r] - mnew);
            float p1 = __expf(s2[1][r] - mnew);
            float ps = p0 + p1;
            #pragma unroll
            for (int off = 1; off <= 8; off <<= 1)
                ps += __shfl_xor(ps, off, 32);
            lrow[r] = lrow[r] * alpha + ps;
            mrow[r] = mnew;
            o0[r] *= alpha; o1[r] *= alpha; o2[r] *= alpha; o3[r] *= alpha;
            const int dl = (r + rbase) + ((r16 & 8) ? 16 : 0);
            const int di = (r16 & 7);
            sPw[dl * 16 + di]     = (_Float16)p0;   // col t*16+r16, t=0
            sPw[dl * 16 + di + 8] = (_Float16)p1;   // t=1
        }

        // P A-fragment: one contiguous 32B read per lane
        v16h pf = *(const v16h*)(sPw + lane * 16);

        // --- O += P V : V tile [key][d] is B's natural row-major layout ---
        #pragma unroll
        for (int j = 0; j < 4; ++j) {
            v16h vf;
            #pragma unroll
            for (int s = 0; s < 16; ++s) vf[s] = sVb[(size_t)kmap(s, lane) * HDIM + j * 16 + r16];
            v8f& oj = (j == 0) ? o0 : (j == 1) ? o1 : (j == 2) ? o2 : o3;
            oj = wmma_f16(pf, vf, oj);
        }
    }

    // epilogue: normalize, scatter to [B, L, DIM] f16 for the out-proj
    #pragma unroll
    for (int r = 0; r < 8; ++r) {
        float inv = 1.0f / lrow[r];
        int q = qt * 16 + r + rbase;
        size_t base = ((size_t)(b * QLEN + q)) * DIM + (size_t)h * HDIM;
        Oh[base +  0 + r16] = (_Float16)(o0[r] * inv);
        Oh[base + 16 + r16] = (_Float16)(o1[r] * inv);
        Oh[base + 32 + r16] = (_Float16)(o2[r] * inv);
        Oh[base + 48 + r16] = (_Float16)(o3[r] * inv);
    }
}

extern "C" void kernel_launch(void* const* d_in, const int* in_sizes, int n_in,
                              void* d_out, int out_size, void* d_ws, size_t ws_size,
                              hipStream_t stream) {
    const float* q_in = (const float*)d_in[0];
    const float* k_in = (const float*)d_in[1];
    const float* v_in = (const float*)d_in[2];
    const unsigned char* mask = (const unsigned char*)d_in[3];
    const float* Wq = (const float*)d_in[4];
    const float* bq = (const float*)d_in[5];
    const float* Wk = (const float*)d_in[6];
    const float* bk = (const float*)d_in[7];
    const float* Wv = (const float*)d_in[8];
    const float* bv = (const float*)d_in[9];
    const float* Wo = (const float*)d_in[10];
    const float* bo = (const float*)d_in[11];

    const int M = BS * QLEN;                   // 8192
    const size_t mat = (size_t)M * DIM;
    _Float16* Qh = (_Float16*)d_ws;            // [B,H,L,64] head-major
    _Float16* Kh = Qh + mat;
    _Float16* Vh = Kh + mat;
    _Float16* Oh = Vh + mat;                   // [B,L,DIM]

    dim3 blk(256);
    dim3 gemm_grid((M / 128) * (DIM / 16));    // 4096 blocks, 8 waves each

    qkv_proj_kernel<<<gemm_grid, blk, 0, stream>>>(q_in, Wq, bq, Qh, M, DIM, DIM, 1);
    qkv_proj_kernel<<<gemm_grid, blk, 0, stream>>>(k_in, Wk, bk, Kh, M, DIM, DIM, 1);
    qkv_proj_kernel<<<gemm_grid, blk, 0, stream>>>(v_in, Wv, bv, Vh, M, DIM, DIM, 1);

    dim3 attn_grid(BS * NHEAD * (QLEN / 128)); // 1024 blocks
    flash_attn_kernel<<<attn_grid, blk, 0, stream>>>(Qh, Kh, Vh, mask, Oh);

    out_proj_kernel<<<gemm_grid, blk, 0, stream>>>(Oh, Wo, bo, (float*)d_out, M, DIM, DIM);
}
